// W8A16Linear_30932354466335
// MI455X (gfx1250) — compile-verified, matrix operand order checked by reading, not run on hardware
//
#include <hip/hip_runtime.h>

// W8A16 GEMM for gfx1250 (MI455X): out = A[M,K](f16) x dequant(int8 W[K,N]) + bias
// 128x128 tile / 256 threads (8 wave32), BK=64, v_wmma_f32_16x16x32_f16, f32 accum.
// A tile moved global->LDS by the Tensor Data Mover (TDM) with HW padding;
// W tile dequantized int8->f16 via v_perm/pk_f16 magic-number trick, stored K-major to LDS.
// Double-buffered LDS, one barrier per K-step, TENSORcnt for TDM completion.
// Output stored with non-temporal hint to preserve L2 residency of Q (64MB < 192MB L2).

typedef __attribute__((ext_vector_type(16))) _Float16 v16h;
typedef __attribute__((ext_vector_type(8)))  _Float16 v8h;
typedef __attribute__((ext_vector_type(4)))  _Float16 v4h;
typedef __attribute__((ext_vector_type(2)))  _Float16 v2h;
typedef __attribute__((ext_vector_type(8)))  float    v8f;
typedef __attribute__((ext_vector_type(4)))  unsigned int v4u;
typedef __attribute__((ext_vector_type(8)))  int      v8i;
typedef __attribute__((ext_vector_type(4)))  int      v4i;

#define BM 128
#define BN 128
#define BK 64
#define LDT 72   // padded LDS row stride in halves (BK + 8): 144B rows -> conflict-free b128 frag loads

#if __has_builtin(__builtin_amdgcn_tensor_load_to_lds)
#define USE_TDM 1
#else
#define USE_TDM 0
#endif

static __device__ __forceinline__ v8f wmma_f16(v8h alo, v8h ahi, v8h blo, v8h bhi, v8f c) {
  v16h a = __builtin_shufflevector(alo, ahi, 0,1,2,3,4,5,6,7,8,9,10,11,12,13,14,15);
  v16h b = __builtin_shufflevector(blo, bhi, 0,1,2,3,4,5,6,7,8,9,10,11,12,13,14,15);
  // (neg_a, A, neg_b, B, c_mod, C, reuse_a, reuse_b)
  return __builtin_amdgcn_wmma_f32_16x16x32_f16(false, a, false, b, (short)0, c, false, false);
}

#if USE_TDM
// TDM: DMA a BM x BK f16 tile (row stride K elements) from global to LDS,
// adding 4 DWORDs of padding after every 32 DWORDs (=> 144B LDS row stride = LDT).
static __device__ __forceinline__ void tdm_load_A(const _Float16* gsrc, _Float16* ldst,
                                                  int K, int M) {
  unsigned lds_off = (unsigned)(unsigned long long)(void*)ldst;          // addr[31:0] = LDS offset
  unsigned long long ga = (unsigned long long)(const void*)gsrc;
  v4u g0;
  g0[0] = 1u;                                                // count=1, user descriptor
  g0[1] = lds_off;                                           // lds_addr
  g0[2] = (unsigned)(ga & 0xffffffffu);                      // global_addr[31:0]
  g0[3] = (unsigned)((ga >> 32) & 0x01ffffffu) | (2u << 30); // global_addr[56:32] | type=2
  v8i g1;
  // w0: wg_mask=0 | data_size=1(2B)<<16 | pad_enable<<20 | pad_interval=4(32DW)<<22 | pad_amount=3(4DW)<<25
  g1[0] = (int)((1u << 16) | (1u << 20) | (4u << 22) | (3u << 25));
  unsigned td0 = (unsigned)K;    // tensor_dim0 (elements along k)
  unsigned td1 = (unsigned)M;    // tensor_dim1 (rows)
  g1[1] = (int)((td0 & 0xffffu) << 16);                        // [47:32]=atomic_barrier=0, [63:48]=td0.lo
  g1[2] = (int)(((td0 >> 16) & 0xffffu) | ((td1 & 0xffffu) << 16));      // td0.hi | td1.lo
  g1[3] = (int)(((td1 >> 16) & 0xffffu) | ((unsigned)BK << 16));         // td1.hi | tile_dim0=64
  g1[4] = (int)((unsigned)BM);                                 // tile_dim1=128 | tile_dim2=0
  g1[5] = (int)((unsigned)K);                                  // tensor_dim0_stride[31:0]
  g1[6] = 0;                                                   // stride[47:32]=0 | dim1_stride.lo=0
  g1[7] = 0;
  v4i gz4 = {0, 0, 0, 0};
  v8i gz8 = {0, 0, 0, 0, 0, 0, 0, 0};
  // 6-arg form (clang-23 / therock-10.0): (g0, g1, g2, g3, g_extra, cpol)
  __builtin_amdgcn_tensor_load_to_lds(g0, g1, gz4, gz4, gz8, 0);
}
#endif

__global__ __launch_bounds__(256)
void w8a16_wmma_kernel(const _Float16* __restrict__ A,
                       const signed char* __restrict__ Q,
                       const _Float16* __restrict__ scales,
                       const _Float16* __restrict__ bias,
                       _Float16* __restrict__ out,
                       int M, int N, int K)
{
  __shared__ __align__(16) _Float16 lA[2][BM * LDT];  // A tile, row-major [m][k], double-buffered
  __shared__ __align__(16) _Float16 lB[2][BN * LDT];  // W tile, K-major   [n][k], dequantized

  const int tid  = threadIdx.x;
  const int lane = tid & 31;
  const int wid  = tid >> 5;
  const int wm   = wid & 3;      // 4 row bands of 32 rows
  const int wn   = wid >> 2;     // 2 col bands of 64 cols
  const int grp  = lane >> 4;    // fragment half-wave group
  const int rl   = lane & 15;    // fragment row/col within 16

  const int n0 = blockIdx.x * BN;
  const int m0 = blockIdx.y * BM;

#if !USE_TDM
  const int a_r0 = tid >> 3;           // A fallback loader: 4 x v8h per thread
  const int a_c  = (tid & 7) * 8;
  v8h aReg[4];
#endif

  // W staging: thread handles 4 k-rows x 8 n-cols of int8
  const int b_n = (tid & 15) * 8;      // 0..120
  const int b_k = (tid >> 4) * 4;      // 0..60

  // Per-output-channel scales (invariant across the K loop): splat to packed f16 pairs.
  v2h sc2[8];
  {
    v8h sv = *reinterpret_cast<const v8h*>(scales + n0 + b_n);
#pragma unroll
    for (int j = 0; j < 8; ++j) { sc2[j][0] = sv[j]; sc2[j][1] = sv[j]; }
  }
  const v2h mm1152 = {(_Float16)(-1152.0f), (_Float16)(-1152.0f)};

  v8f acc[2][4];
#pragma unroll
  for (int i = 0; i < 2; ++i)
#pragma unroll
    for (int j = 0; j < 4; ++j)
#pragma unroll
      for (int r = 0; r < 8; ++r) acc[i][j][r] = 0.0f;

  const int nK = K / BK;
  unsigned qx[4][2];   // int8 dwords, pre-XORed with 0x80 bias

  // ---- helpers ----
  auto loadW = [&](int kbase) {
#pragma unroll
    for (int kk = 0; kk < 4; ++kk) {
      unsigned long long q = *reinterpret_cast<const unsigned long long*>(
          Q + (size_t)(kbase + b_k + kk) * N + n0 + b_n);
      qx[kk][0] = (unsigned)q ^ 0x80808080u;
      qx[kk][1] = (unsigned)(q >> 32) ^ 0x80808080u;
    }
  };
  // int8 -> f16 dequant, 4 k-values per column j packed with v_perm + pk_f16 magic:
  // half = (0x6400 | (q^0x80)) - 1152.0h == (f16)q exactly; then * scale (f16 RNE, matches ref)
  auto stageW = [&](int buf) {
#pragma unroll
    for (int h = 0; h < 2; ++h) {
#pragma unroll
      for (int jj = 0; jj < 4; ++jj) {
        const int j = h * 4 + jj;
        const unsigned sel = 0x0C000C00u | (unsigned)jj | ((unsigned)(4 + jj) << 16);
        unsigned p01 = __builtin_amdgcn_perm(qx[1][h], qx[0][h], sel) | 0x64006400u;
        unsigned p23 = __builtin_amdgcn_perm(qx[3][h], qx[2][h], sel) | 0x64006400u;
        v2h w01 = (__builtin_bit_cast(v2h, p01) + mm1152) * sc2[j];
        v2h w23 = (__builtin_bit_cast(v2h, p23) + mm1152) * sc2[j];
        v4h w = __builtin_shufflevector(w01, w23, 0, 1, 2, 3);
        *reinterpret_cast<v4h*>(&lB[buf][(b_n + j) * LDT + b_k]) = w;
      }
    }
  };
#if !USE_TDM
  auto loadA = [&](int kbase) {
#pragma unroll
    for (int it = 0; it < 4; ++it)
      aReg[it] = *reinterpret_cast<const v8h*>(A + (size_t)(m0 + a_r0 + it * 32) * K + kbase + a_c);
  };
  auto stageA = [&](int buf) {
#pragma unroll
    for (int it = 0; it < 4; ++it)
      *reinterpret_cast<v8h*>(&lA[buf][(a_r0 + it * 32) * LDT + a_c]) = aReg[it];
  };
#endif

  // ---- prologue: stage tile 0 into buffer 0 ----
#if USE_TDM
  if (wid == 0) tdm_load_A(A + (size_t)m0 * K, &lA[0][0], K, M);
#else
  loadA(0);
#endif
  loadW(0);
  stageW(0);
#if !USE_TDM
  stageA(0);
#else
  if (wid == 0) __builtin_amdgcn_s_wait_tensorcnt(0);
#endif
  __syncthreads();

  // ---- main loop: one barrier per K-step, double-buffered LDS ----
  for (int kt = 0; kt < nK; ++kt) {
    const int c = kt & 1;
    const bool more = (kt + 1 < nK);

    if (more) {
      const int kn = (kt + 1) * BK;
#if USE_TDM
      if (wid == 0) tdm_load_A(A + (size_t)m0 * K + kn, &lA[1 - c][0], K, M);
#else
      loadA(kn);
#endif
      loadW(kn);
    }

    // compute: 2 sub-steps of K=32 -> 16 WMMAs per K-tile per wave
#pragma unroll
    for (int kk = 0; kk < 2; ++kk) {
      const int ks = kk * 32;
      v8h alo[2], ahi[2];
#pragma unroll
      for (int i = 0; i < 2; ++i) {
        const _Float16* p = &lA[c][(wm * 32 + i * 16 + rl) * LDT + ks + grp * 8];
        alo[i] = *reinterpret_cast<const v8h*>(p);
        ahi[i] = *reinterpret_cast<const v8h*>(p + 16);
      }
      v8h blo[4], bhi[4];
#pragma unroll
      for (int j = 0; j < 4; ++j) {
        const _Float16* p = &lB[c][(wn * 64 + j * 16 + rl) * LDT + ks + grp * 8];
        blo[j] = *reinterpret_cast<const v8h*>(p);
        bhi[j] = *reinterpret_cast<const v8h*>(p + 16);
      }
#pragma unroll
      for (int i = 0; i < 2; ++i)
#pragma unroll
        for (int j = 0; j < 4; ++j)
          acc[i][j] = wmma_f16(alo[i], ahi[i], blo[j], bhi[j], acc[i][j]);
    }

    if (more) {
      stageW(1 - c);
#if !USE_TDM
      stageA(1 - c);
#else
      if (wid == 0) __builtin_amdgcn_s_wait_tensorcnt(0);
#endif
      __syncthreads();   // publishes buffer 1-c; also guarantees buffer c drained before next overwrite
    }
  }

  // ---- epilogue: bias add (f32), fp16 NT store per C/D layout (VGPR r -> M=grp*8+r, lane%16 -> N) ----
#pragma unroll
  for (int j = 0; j < 4; ++j) {
    const int col = n0 + wn * 64 + j * 16 + rl;
    const float bj = (float)bias[col];
#pragma unroll
    for (int i = 0; i < 2; ++i) {
      const int row0 = m0 + wm * 32 + i * 16 + grp * 8;
#pragma unroll
      for (int r = 0; r < 8; ++r)
        __builtin_nontemporal_store((_Float16)(acc[i][j][r] + bj),
                                    &out[(size_t)(row0 + r) * N + col]);
    }
  }
}

extern "C" void kernel_launch(void* const* d_in, const int* in_sizes, int n_in,
                              void* d_out, int out_size, void* d_ws, size_t ws_size,
                              hipStream_t stream) {
  const _Float16*    A      = (const _Float16*)d_in[0];     // fp16 [M,K]
  const signed char* Q      = (const signed char*)d_in[1];  // int8 [K,N]
  const _Float16*    scales = (const _Float16*)d_in[2];     // fp16 [N]
  const _Float16*    bias   = (const _Float16*)d_in[3];     // fp16 [N]
  _Float16*          out    = (_Float16*)d_out;             // fp16 [M,N]

  const int N = in_sizes[2];
  const int K = in_sizes[1] / N;
  const int M = in_sizes[0] / K;

  dim3 grid(N / BN, M / BM);   // sizes are multiples of 128 for this problem
  w8a16_wmma_kernel<<<grid, 256, 0, stream>>>(A, Q, scales, bias, out, M, N, K);
}
